// NNUE_9148280341053
// MI455X (gfx1250) — matrix operand order repaired and checked by reading, not compiled, not run on hardware
//
#include <hip/hip_runtime.h>

typedef __attribute__((ext_vector_type(16))) __bf16 v16bf;
typedef __attribute__((ext_vector_type(8)))  __bf16 v8bf;
typedef __attribute__((ext_vector_type(4)))  __bf16 v4bf;
typedef __attribute__((ext_vector_type(8)))  float  v8f;
typedef __attribute__((ext_vector_type(4)))  float  v4f;

#define FDIM   40960
#define BATCH  4096
#define MDIM   256
#define LDA_ST 40   // padded LDS stride (elements) -> 80B, conflict-free b128 frag reads

// ---------------------------------------------------------------------------
// Kernel 1: convert ft_w [256, 40960] fp32 -> bf16 (row-major, same layout).
// ---------------------------------------------------------------------------
__global__ __launch_bounds__(256)
void nnue_cvt_w(const float* __restrict__ w, __bf16* __restrict__ wb) {
    const size_t i = ((size_t)blockIdx.x * 256 + threadIdx.x) * 8;
    v4f f0 = *(const v4f*)(w + i);
    v4f f1 = *(const v4f*)(w + i + 4);
    v8bf o;
    o[0] = (__bf16)f0[0]; o[1] = (__bf16)f0[1]; o[2] = (__bf16)f0[2]; o[3] = (__bf16)f0[3];
    o[4] = (__bf16)f1[0]; o[5] = (__bf16)f1[1]; o[6] = (__bf16)f1[2]; o[7] = (__bf16)f1[3];
    *(v8bf*)(wb + i) = o;
}

// ---------------------------------------------------------------------------
// Kernel 2: feature-transformer GEMM  acc[B,256] = X[B,40960] @ ft_w^T
//   Block tile: 32 rows x 256 cols. 8 waves = 2 row-subtiles x 4 col-quarters;
//   each wave: 4 x v_wmma_f32_16x16x32_bf16 per K-step (32 acc VGPRs -> low
//   register pressure, no accumulator spill/copies).
//   Double-buffered LDS; B tile staged with global_load_async_to_lds_b128
//   (ASYNCcnt), A tile converted fp32->bf16 in regs then ds_store'd.
// ---------------------------------------------------------------------------
__global__ __launch_bounds__(256)
void nnue_ft_gemm(const float* __restrict__ white, const float* __restrict__ black,
                  const __bf16* __restrict__ Wb,
                  float* __restrict__ wacc, float* __restrict__ bacc) {
    __shared__ __align__(16) __bf16 lA[2][32 * LDA_ST];    // 2 x  2.5 KB
    __shared__ __align__(16) __bf16 lB[2][256 * LDA_ST];   // 2 x 20.0 KB

    const float* __restrict__ X  = (blockIdx.y == 0) ? white : black;
    float* __restrict__ acc_out  = (blockIdx.y == 0) ? wacc  : bacc;

    const int tid      = threadIdx.x;
    const int lane     = tid & 31;
    const int wave     = tid >> 5;
    const int row_sub  = wave & 1;   // which 16-row sub-tile (of 32)
    const int col_q    = wave >> 1;  // which 64-col quarter (of 256)
    const int r0       = blockIdx.x * 32;

    // fragment lane decomposition (wave32)
    const int fr = lane & 15;              // A: M row / B: N col within 16x16 tile
    const int ka = (lane >> 4) * 8;        // A frag: K chunks {ka..ka+7, ka+16..ka+23}
    const int kb = (lane >> 4) * 8;        // B frag: K chunks {kb..kb+7, kb+16..kb+23}

    v8f acc[4];
    #pragma unroll
    for (int c = 0; c < 4; ++c)
        #pragma unroll
        for (int i = 0; i < 8; ++i) acc[c][i] = 0.0f;

    // cooperative staging indices: A tile is 32x32 fp32 = one v4f per thread
    const int a_row = tid >> 3;            // 0..31
    const int a_k   = (tid & 7) * 4;       // 0,4,...,28
    const float*  __restrict__ aptr = X  + (size_t)(r0 + a_row) * FDIM + a_k;
    const __bf16* __restrict__ bptr = Wb + (size_t)tid * FDIM;   // Wb row tid (0..255)

    // per-thread LDS byte offsets (low 32 bits of flat shared address == LDS offset)
    const unsigned ldsB0 = (unsigned)(size_t)(void*)&lB[0][tid * LDA_ST];
    const unsigned ldsB1 = (unsigned)(size_t)(void*)&lB[1][tid * LDA_ST];

    // ---- staging helpers -------------------------------------------------
    auto stage_a = [&](int buf, int k) {
        v4f f0 = *(const v4f*)(aptr + k);
        __builtin_prefetch(aptr + k + 512, 0, 1);   // stream hint for X
        v4bf ab;
        ab[0] = (__bf16)f0[0]; ab[1] = (__bf16)f0[1];
        ab[2] = (__bf16)f0[2]; ab[3] = (__bf16)f0[3];
        *(v4bf*)(&lA[buf][a_row * LDA_ST + a_k]) = ab;
    };
    auto stage_b_async = [&](int buf, int k) {
        const unsigned lds = buf ? ldsB1 : ldsB0;     // 64B per row, contiguous
        const __bf16* g = bptr + k;
        asm volatile(
            "global_load_async_to_lds_b128 %0, %1, off\n\t"
            "global_load_async_to_lds_b128 %0, %1, off offset:16\n\t"
            "global_load_async_to_lds_b128 %0, %1, off offset:32\n\t"
            "global_load_async_to_lds_b128 %0, %1, off offset:48"
            :: "v"(lds), "v"(g) : "memory");
    };
    auto compute = [&](int buf) {
        const __bf16* arow = &lA[buf][(row_sub * 16 + fr) * LDA_ST];
        v8bf alo = *(const v8bf*)(arow + ka);
        v8bf ahi = *(const v8bf*)(arow + ka + 16);
        v16bf afrag = __builtin_shufflevector(alo, ahi,
                        0,1,2,3,4,5,6,7,8,9,10,11,12,13,14,15);
        #pragma unroll
        for (int c = 0; c < 4; ++c) {
            const __bf16* brow = &lB[buf][(col_q * 64 + c * 16 + fr) * LDA_ST];
            v8bf blo = *(const v8bf*)(brow + kb);
            v8bf bhi = *(const v8bf*)(brow + kb + 16);
            v16bf bfrag = __builtin_shufflevector(blo, bhi,
                            0,1,2,3,4,5,6,7,8,9,10,11,12,13,14,15);
            acc[c] = __builtin_amdgcn_wmma_f32_16x16x32_bf16(
                         false, afrag, false, bfrag,
                         (short)0, acc[c], false, false);
        }
    };

    // ---- prologue: fill buffer 0 ----------------------------------------
    stage_a(0, 0);
    stage_b_async(0, 0);
    asm volatile("s_wait_asynccnt 0x0" ::: "memory");
    __syncthreads();

    // ---- main loop: single compute site, stage next while computing -----
    int buf = 0;
    for (int k = 0; k < FDIM; k += 32) {
        const int kn = k + 32;
        if (kn < FDIM) {
            stage_a(buf ^ 1, kn);        // global f32 loads + cvt + ds_store
            stage_b_async(buf ^ 1, kn);  // async global->LDS, no VGPR data
        }
        compute(buf);                    // 4 x WMMA from current buf
        asm volatile("s_wait_asynccnt 0x0" ::: "memory");
        __syncthreads();
        buf ^= 1;
    }

    // ---- epilogue: C layout -> VGPR i holds M = i (+8 for lanes 16..31), N = lane%16
    const int mrow = (lane < 16) ? 0 : 8;
    float* obase = acc_out + (size_t)(r0 + row_sub * 16) * MDIM
                           + col_q * 64 + (lane & 15);
    #pragma unroll
    for (int c = 0; c < 4; ++c)
        #pragma unroll
        for (int i = 0; i < 8; ++i)
            obase[(size_t)(mrow + i) * MDIM + c * 16] = acc[c][i];
}

// ---------------------------------------------------------------------------
// Kernel 3: tail. One wave per batch row:
//   x = clip(stm-selected concat(w,b) + ft_b), h = clip(x @ l1_w^T + l1_b),
//   out = h @ l2_w^T + l2_b
// ---------------------------------------------------------------------------
__global__ __launch_bounds__(256)
void nnue_tail(const float* __restrict__ wacc, const float* __restrict__ bacc,
               const float* __restrict__ stm,  const float* __restrict__ ft_b,
               const float* __restrict__ l1_w, const float* __restrict__ l1_b,
               const float* __restrict__ l2_w, const float* __restrict__ l2_b,
               float* __restrict__ out) {
    __shared__ float xbuf[8][512];
    const int lane = threadIdx.x & 31;
    const int wave = threadIdx.x >> 5;
    const int b    = blockIdx.x * 8 + wave;

    const float s = stm[b];
    const float* first  = (s > 0.5f) ? (wacc + (size_t)b * MDIM) : (bacc + (size_t)b * MDIM);
    const float* second = (s > 0.5f) ? (bacc + (size_t)b * MDIM) : (wacc + (size_t)b * MDIM);

    #pragma unroll
    for (int i = 0; i < 8; ++i) {
        const int j = lane * 8 + i;                 // 0..255
        const float bias = ft_b[j];
        xbuf[wave][j]       = fminf(fmaxf(first[j]  + bias, 0.0f), 1.0f);
        xbuf[wave][256 + j] = fminf(fmaxf(second[j] + bias, 0.0f), 1.0f);
    }
    __syncthreads();

    // lane n computes hidden unit n (l1_w row is contiguous, L2 resident)
    const float* wrow = l1_w + lane * 512;
    float h = l1_b[lane];
    #pragma unroll 4
    for (int j = 0; j < 512; j += 4) {
        v4f wv = *(const v4f*)(wrow + j);
        h += xbuf[wave][j]     * wv[0] + xbuf[wave][j + 1] * wv[1]
           + xbuf[wave][j + 2] * wv[2] + xbuf[wave][j + 3] * wv[3];
    }
    h = fminf(fmaxf(h, 0.0f), 1.0f);

    float v = h * l2_w[lane];
    #pragma unroll
    for (int off = 16; off > 0; off >>= 1)
        v += __shfl_xor(v, off, 32);
    if (lane == 0) out[b] = v + l2_b[0];
}

// ---------------------------------------------------------------------------
extern "C" void kernel_launch(void* const* d_in, const int* in_sizes, int n_in,
                              void* d_out, int out_size, void* d_ws, size_t ws_size,
                              hipStream_t stream) {
    const float* white = (const float*)d_in[0];
    const float* black = (const float*)d_in[1];
    const float* stm   = (const float*)d_in[2];
    const float* ft_w  = (const float*)d_in[3];
    const float* ft_b  = (const float*)d_in[4];
    const float* l1_w  = (const float*)d_in[5];
    const float* l1_b  = (const float*)d_in[6];
    const float* l2_w  = (const float*)d_in[7];
    const float* l2_b  = (const float*)d_in[8];
    float* out = (float*)d_out;

    // workspace: Wb (bf16 weights, 20.97 MB) | wacc (4.19 MB) | bacc (4.19 MB)
    char*   ws   = (char*)d_ws;
    __bf16* Wb   = (__bf16*)ws;
    float*  wacc = (float*)(ws + (size_t)MDIM * FDIM * sizeof(__bf16));
    float*  bacc = wacc + (size_t)BATCH * MDIM;

    // 1) ft_w fp32 -> bf16 (L2-resident weight copy)
    nnue_cvt_w<<<(MDIM * FDIM) / (256 * 8), 256, 0, stream>>>(ft_w, Wb);

    // 2) feature transformer: both perspectives, 32-row tiles
    nnue_ft_gemm<<<dim3(BATCH / 32, 2), 256, 0, stream>>>(white, black, Wb, wacc, bacc);

    // 3) tail MLP
    nnue_tail<<<BATCH / 8, 256, 0, stream>>>(wacc, bacc, stm, ft_b,
                                             l1_w, l1_b, l2_w, l2_b, out);
}
